// OrderAttention_27839978013188
// MI455X (gfx1250) — compile-verified, matrix operand order checked by reading, not run on hardware
//
#include <hip/hip_runtime.h>

#define DIMX 1024
#define HEADS 16
#define DHEAD 64
#define BATCH 4
#define SEQ 2048
#define INNER 1024
#define OUT3 3072
#define SCALE 0.125f
#define NEGV -1e9f

typedef __attribute__((ext_vector_type(16))) __bf16 v16bf;
typedef __attribute__((ext_vector_type(8)))  float  v8f;

union FragU { uint4 q[2]; v16bf v; unsigned short s[16]; };

__device__ __forceinline__ unsigned short f2bf(float f) {
  unsigned u = __float_as_uint(f);
  u += 0x7FFFu + ((u >> 16) & 1u);   // round-to-nearest-even
  return (unsigned short)(u >> 16);
}

// Load a 16x32 bf16 operand fragment from a row-major [16 x ld] tile.
// Lane L<16: row L, K = {0..7, 16..23}; lane L>=16: row L-16, K = {8..15, 24..31}.
// Works for both the A operand and (applied to B^T rows) the B operand.
__device__ __forceinline__ v16bf load_frag(const unsigned short* base, int ld) {
  int lane = threadIdx.x & 31;
  int grp  = lane >> 4;
  int row  = lane & 15;
  const uint4* p = (const uint4*)(base + row * ld + (grp << 3));
  FragU u;
  u.q[0] = p[0];   // K = 8g .. 8g+7
  u.q[1] = p[2];   // K = 16+8g .. 16+8g+7  (16 elems = 32B ahead)
  return u.v;
}

__device__ __forceinline__ v8f wmma_bf16(v16bf a, v16bf b, v8f c) {
  return __builtin_amdgcn_wmma_f32_16x16x32_bf16(false, a, false, b, (short)0, c,
                                                 false, false);
}

// ---------------- prep kernels ----------------
__global__ __launch_bounds__(256) void cvt_x_kernel(const float* __restrict__ x,
                                                    unsigned short* __restrict__ xb,
                                                    int n) {
  int i = blockIdx.x * blockDim.x + threadIdx.x;
  if (i < n) xb[i] = f2bf(x[i]);
}

// w: [rows][cols] fp32 -> wt: [cols][rows] bf16
__global__ __launch_bounds__(256) void cvt_t_kernel(const float* __restrict__ w,
                                                    unsigned short* __restrict__ wt,
                                                    int rows, int cols) {
  int i = blockIdx.x * blockDim.x + threadIdx.x;
  if (i < rows * cols) {
    int o = i / rows;
    int r = i - o * rows;
    wt[i] = f2bf(w[r * cols + o]);
  }
}

// ---------------- QKV GEMM: [8192,1024] @ [1024,3072] ----------------
// One wave -> 32x64 output tile. Scatters q,k row-major and v transposed.
__global__ __launch_bounds__(256) void qkv_gemm_kernel(
    const unsigned short* __restrict__ xb,
    const unsigned short* __restrict__ wT,   // [3072][1024]
    unsigned short* __restrict__ qb,         // [b*h][n][64]
    unsigned short* __restrict__ kb,         // [b*h][n][64]
    unsigned short* __restrict__ vT) {       // [b*h][64][n]
  int wave = (blockIdx.x * blockDim.x + threadIdx.x) >> 5;
  const int NT = OUT3 / 64;                  // 48
  int mt = wave / NT, nt = wave - mt * NT;
  int i0 = mt * 32, o0 = nt * 64;

  v8f c[2][4];
#pragma unroll
  for (int m = 0; m < 2; ++m)
#pragma unroll
    for (int n = 0; n < 4; ++n) c[m][n] = (v8f){0.f,0.f,0.f,0.f,0.f,0.f,0.f,0.f};

  for (int k0 = 0; k0 < DIMX; k0 += 32) {
    v16bf a0 = load_frag(xb + (size_t)i0 * DIMX + k0, DIMX);
    v16bf a1 = load_frag(xb + (size_t)(i0 + 16) * DIMX + k0, DIMX);
#pragma unroll
    for (int n = 0; n < 4; ++n) {
      v16bf b = load_frag(wT + (size_t)(o0 + 16 * n) * DIMX + k0, DIMX);
      c[0][n] = wmma_bf16(a0, b, c[0][n]);
      c[1][n] = wmma_bf16(a1, b, c[1][n]);
    }
  }

  int lane = threadIdx.x & 31, grp = lane >> 4, col = lane & 15;
  int sec = o0 / INNER;
  int h   = (o0 % INNER) / DHEAD;
  int b   = i0 / SEQ;
  int n0  = i0 % SEQ;
  int bh  = b * HEADS + h;
#pragma unroll
  for (int m = 0; m < 2; ++m)
#pragma unroll
    for (int n = 0; n < 4; ++n)
#pragma unroll
      for (int r = 0; r < 8; ++r) {
        int row = n0 + m * 16 + r + 8 * grp;
        int d   = n * 16 + col;
        unsigned short val = f2bf(c[m][n][r]);
        if (sec == 0)      qb[((size_t)bh * SEQ + row) * DHEAD + d] = val;
        else if (sec == 1) kb[((size_t)bh * SEQ + row) * DHEAD + d] = val;
        else               vT[((size_t)bh * DHEAD + d) * SEQ + row] = val;
      }
}

// ---------------- Flash attention: one wave per 16 query rows ----------------
// S^T = K.Q^T (so exp(S^T) C-frag IS the B operand layout for O^T = V^T.P^T)
__global__ __launch_bounds__(256) void attn_kernel(
    const unsigned short* __restrict__ qb,
    const unsigned short* __restrict__ kb,
    const unsigned short* __restrict__ vT,
    unsigned short* __restrict__ ob) {       // [b][n][1024] bf16
  int wave = (blockIdx.x * blockDim.x + threadIdx.x) >> 5;
  int bh = wave >> 7;                        // 64 (b,h) pairs
  int qt = wave & 127;                       // 128 query tiles
  int qi0 = qt * 16;
  int lane = threadIdx.x & 31;
  int grp = lane >> 4;
  int iCol = lane & 15;
  int iglob = qi0 + iCol;

  const unsigned short* qbase = qb + ((size_t)bh * SEQ + qi0) * DHEAD;
  v16bf qB0 = load_frag(qbase, DHEAD);       // d 0..31
  v16bf qB1 = load_frag(qbase + 32, DHEAD);  // d 32..63

  v8f o0 = {0.f,0.f,0.f,0.f,0.f,0.f,0.f,0.f};
  v8f o1 = o0, o2 = o0, o3 = o0;
  float mrow = -3.0e38f, lrow = 0.0f;

  int nChunks = (qi0 + 16 + 31) >> 5;
  for (int jc = 0; jc < nChunks; ++jc) {
    int jb = jc << 5;
    const unsigned short* kbase = kb + ((size_t)bh * SEQ + jb) * DHEAD;
    v8f s0 = {0.f,0.f,0.f,0.f,0.f,0.f,0.f,0.f};
    v8f s1 = s0;
    s0 = wmma_bf16(load_frag(kbase, DHEAD), qB0, s0);
    s0 = wmma_bf16(load_frag(kbase + 32, DHEAD), qB1, s0);
    s1 = wmma_bf16(load_frag(kbase + 16 * DHEAD, DHEAD), qB0, s1);
    s1 = wmma_bf16(load_frag(kbase + 16 * DHEAD + 32, DHEAD), qB1, s1);

    float p0[8], p1[8];
    float cmax = -3.0e38f;
#pragma unroll
    for (int r = 0; r < 8; ++r) {
      int j0 = jb + r + 8 * grp;             // key index of s0[r]
      int j1 = j0 + 16;                      // key index of s1[r]
      float v0 = (j0 <= iglob) ? s0[r] * SCALE : NEGV;
      float v1 = (j1 <= iglob) ? s1[r] * SCALE : NEGV;
      p0[r] = v0; p1[r] = v1;
      cmax = fmaxf(cmax, fmaxf(v0, v1));
    }
    cmax = fmaxf(cmax, __shfl_xor(cmax, 16, 32));  // combine j-halves
    float mnew = fmaxf(mrow, cmax);
    float alpha = __expf(mrow - mnew);
    float rsum = 0.f;
    FragU pb;
#pragma unroll
    for (int r = 0; r < 8; ++r) {
      float e0 = __expf(p0[r] - mnew);
      float e1 = __expf(p1[r] - mnew);
      rsum += e0 + e1;
      pb.s[r]     = f2bf(e0);                // B-operand k = 8g+r
      pb.s[8 + r] = f2bf(e1);                // B-operand k = 16+8g+r
    }
    rsum += __shfl_xor(rsum, 16, 32);
    lrow = lrow * alpha + rsum;
    mrow = mnew;
#pragma unroll
    for (int r = 0; r < 8; ++r) {
      o0[r] *= alpha; o1[r] *= alpha; o2[r] *= alpha; o3[r] *= alpha;
    }
    const unsigned short* vbase = vT + (size_t)bh * DHEAD * SEQ + jb;
    o0 = wmma_bf16(load_frag(vbase, SEQ), pb.v, o0);
    o1 = wmma_bf16(load_frag(vbase + 16 * SEQ, SEQ), pb.v, o1);
    o2 = wmma_bf16(load_frag(vbase + 32 * SEQ, SEQ), pb.v, o2);
    o3 = wmma_bf16(load_frag(vbase + 48 * SEQ, SEQ), pb.v, o3);
  }

  float rl = 1.0f / lrow;                    // per lane: query row i = lane&15
  int b = bh >> 4, h = bh & 15;
  unsigned short* outp = ob + ((size_t)b * SEQ + iglob) * INNER + h * DHEAD;
#pragma unroll
  for (int r = 0; r < 8; ++r) {
    int d = r + 8 * grp;                     // O^T frag: reg r holds d-row
    outp[d]      = f2bf(o0[r] * rl);
    outp[16 + d] = f2bf(o1[r] * rl);
    outp[32 + d] = f2bf(o2[r] * rl);
    outp[48 + d] = f2bf(o3[r] * rl);
  }
}

// ---------------- Output projection: [8192,1024] @ [1024,1024] + bias ----------------
__global__ __launch_bounds__(256) void out_gemm_kernel(
    const unsigned short* __restrict__ ab,
    const unsigned short* __restrict__ wT,   // [1024][1024] = w_out^T
    const float* __restrict__ bias,
    float* __restrict__ out) {
  int wave = (blockIdx.x * blockDim.x + threadIdx.x) >> 5;
  const int NT = DIMX / 64;                  // 16
  int mt = wave / NT, nt = wave - mt * NT;
  int i0 = mt * 32, o0 = nt * 64;

  v8f c[2][4];
#pragma unroll
  for (int m = 0; m < 2; ++m)
#pragma unroll
    for (int n = 0; n < 4; ++n) c[m][n] = (v8f){0.f,0.f,0.f,0.f,0.f,0.f,0.f,0.f};

  for (int k0 = 0; k0 < INNER; k0 += 32) {
    v16bf a0 = load_frag(ab + (size_t)i0 * INNER + k0, INNER);
    v16bf a1 = load_frag(ab + (size_t)(i0 + 16) * INNER + k0, INNER);
#pragma unroll
    for (int n = 0; n < 4; ++n) {
      v16bf b = load_frag(wT + (size_t)(o0 + 16 * n) * INNER + k0, INNER);
      c[0][n] = wmma_bf16(a0, b, c[0][n]);
      c[1][n] = wmma_bf16(a1, b, c[1][n]);
    }
  }

  int lane = threadIdx.x & 31, grp = lane >> 4, col = lane & 15;
#pragma unroll
  for (int m = 0; m < 2; ++m)
#pragma unroll
    for (int n = 0; n < 4; ++n)
#pragma unroll
      for (int r = 0; r < 8; ++r) {
        int row = i0 + m * 16 + r + 8 * grp;
        int o   = o0 + n * 16 + col;
        out[(size_t)row * DIMX + o] = c[m][n][r] + bias[o];
      }
}

extern "C" void kernel_launch(void* const* d_in, const int* in_sizes, int n_in,
                              void* d_out, int out_size, void* d_ws, size_t ws_size,
                              hipStream_t stream) {
  const float* x     = (const float*)d_in[0];
  const float* w_qkv = (const float*)d_in[1];
  const float* w_out = (const float*)d_in[2];
  const float* b_out = (const float*)d_in[3];
  float* out = (float*)d_out;

  // workspace layout (ushort/bf16 elements); total = 44M elems = 88 MB
  unsigned short* ws    = (unsigned short*)d_ws;
  unsigned short* xb    = ws;                                  // 8M
  unsigned short* wqkvT = xb    + (size_t)8 * 1024 * 1024;     // 3M
  unsigned short* woutT = wqkvT + (size_t)3 * 1024 * 1024;     // 1M
  unsigned short* qb    = woutT + (size_t)1 * 1024 * 1024;     // 8M
  unsigned short* kb    = qb    + (size_t)8 * 1024 * 1024;     // 8M
  unsigned short* vT    = kb    + (size_t)8 * 1024 * 1024;     // 8M
  unsigned short* ab    = vT    + (size_t)8 * 1024 * 1024;     // 8M

  int nx = BATCH * SEQ * DIMX;                                 // 8388608
  cvt_x_kernel<<<nx / 256, 256, 0, stream>>>(x, xb, nx);
  cvt_t_kernel<<<(DIMX * OUT3) / 256, 256, 0, stream>>>(w_qkv, wqkvT, DIMX, OUT3);
  cvt_t_kernel<<<(DIMX * DIMX) / 256, 256, 0, stream>>>(w_out, woutT, DIMX, DIMX);
  qkv_gemm_kernel<<<(256 * 48) / 8, 256, 0, stream>>>(xb, wqkvT, qb, kb, vT);
  attn_kernel<<<(64 * 128) / 8, 256, 0, stream>>>(qb, kb, vT, ab);
  out_gemm_kernel<<<(256 * 16) / 8, 256, 0, stream>>>(ab, woutT, b_out, out);
}
